// manual_SDPA_56186762166562
// MI455X (gfx1250) — matrix operand order repaired and checked by reading, not compiled, not run on hardware
//
#include <hip/hip_runtime.h>

#define B_ 16
#define S_ 2048
#define D_ 128
#define QT 128      // q rows per workgroup
#define KC 32       // keys per iteration (fallback kernel)
#define KC2 64      // keys per iteration (bf16-workspace kernel)
#define NW 8        // waves per workgroup (wave32)

// scale = (1/sqrt(D)) * log2(e): lets us use native v_exp_f32 (2^x) directly
#define QSCALE (0.08838834764831845f * 1.4426950408889634f)

typedef __bf16 v16bf  __attribute__((ext_vector_type(16)));
typedef __bf16 bf16x8 __attribute__((ext_vector_type(8)));
typedef float  v8f    __attribute__((ext_vector_type(8)));
typedef int    v4i_   __attribute__((ext_vector_type(4)));

union Frag16 { v16bf v; bf16x8 h[2]; };

#if __has_builtin(__builtin_amdgcn_global_load_async_to_lds_b128)
#define HAVE_ASYNC 1
#else
#define HAVE_ASYNC 0
#endif

__device__ __forceinline__ __bf16 f2bf(float f) {
  union { float f; unsigned u; } x; x.f = f;
  unsigned r = x.u + 0x7FFFu + ((x.u >> 16) & 1u);   // round-to-nearest-even
  union { unsigned short s; __bf16 b; } y; y.s = (unsigned short)(r >> 16);
  return y.b;
}

__device__ __forceinline__ float exp2_native(float x) {
#if __has_builtin(__builtin_amdgcn_exp2f)
  return __builtin_amdgcn_exp2f(x);    // bare v_exp_f32, no denorm guard
#else
  return exp2f(x);
#endif
}

// bf16 truncation store: high 16 bits of the f32 -> matches ds_store_b16_d16_hi
__device__ __forceinline__ void store_bf16_hi(__bf16* p, float f) {
  *(unsigned short*)p = (unsigned short)(__float_as_uint(f) >> 16);
}

__device__ __forceinline__ void wait_async0() {
#if __has_builtin(__builtin_amdgcn_s_wait_asynccnt)
  __builtin_amdgcn_s_wait_asynccnt(0);
#else
  asm volatile("s_wait_asynccnt 0" ::: "memory");
#endif
}

__device__ __forceinline__ void cp16(const __bf16* g, __bf16* l) {
#if HAVE_ASYNC
  __builtin_amdgcn_global_load_async_to_lds_b128(
      (__attribute__((address_space(1))) v4i_*)g,
      (__attribute__((address_space(3))) v4i_*)l, 0, 0);
#else
  *(uint4*)l = *(const uint4*)g;
#endif
}

// ---------------------------------------------------------------------------
// Fallback kernel (f32 inputs, KC=32) — used when ws is too small.
// ---------------------------------------------------------------------------
__global__ void __launch_bounds__(256)
attn_pass1(const float* __restrict__ q, const float* __restrict__ k,
           const float* __restrict__ v, float* __restrict__ out,
           float* __restrict__ total) {
  __shared__ __align__(16) __bf16 q_lds[QT * D_];
  __shared__ __align__(16) __bf16 k_lds[KC * D_];
  __shared__ __align__(16) __bf16 vt_lds[D_ * KC];
  __shared__ __align__(16) __bf16 p_lds[NW][16 * KC];

  const int tid  = threadIdx.x;
  const int w    = tid >> 5;
  const int lane = tid & 31;
  const int ln   = lane & 15;
  const int hi   = lane >> 4;

  const int b  = blockIdx.y;
  const int q0 = blockIdx.x * QT;
  const size_t bbase = (size_t)b * S_ * D_;

  for (int e = tid; e < QT * D_; e += 256) {
    const int row = e >> 7, col = e & (D_ - 1);
    q_lds[e] = f2bf(q[bbase + (size_t)(q0 + row) * D_ + col] * QSCALE);
  }
  __syncthreads();

  Frag16 qa[4];
  {
    const int m = w * 16 + ln;
    #pragma unroll
    for (int c = 0; c < 4; ++c) {
      const int kb = c * 32 + hi * 8;
      qa[c].h[0] = *(const bf16x8*)&q_lds[m * D_ + kb];
      qa[c].h[1] = *(const bf16x8*)&q_lds[m * D_ + kb + 16];
    }
  }

  v8f o[8];
  #pragma unroll
  for (int g = 0; g < 8; ++g) o[g] = (v8f)(0.0f);
  float sumexp = 0.0f;

  for (int k0 = 0; k0 < S_; k0 += KC) {
    __syncthreads();
    for (int e = tid; e < KC * D_; e += 256) {
      const int row = e >> 7, col = e & (D_ - 1);
      const size_t ga = bbase + (size_t)(k0 + row) * D_ + col;
      k_lds[e] = f2bf(k[ga]);
      vt_lds[col * KC + row] = f2bf(v[ga]);
    }
    __syncthreads();

    #pragma unroll
    for (int f = 0; f < 2; ++f) {
      v8f s = (v8f)(0.0f);
      const int krow = f * 16 + ln;
      #pragma unroll
      for (int c = 0; c < 4; ++c) {
        Frag16 kb;
        const bf16x8* kp = (const bf16x8*)&k_lds[krow * D_ + c * 32 + hi * 8];
        kb.h[0] = kp[0]; kb.h[1] = kp[2];
        s = __builtin_amdgcn_wmma_f32_16x16x32_bf16(false, qa[c].v, false, kb.v,
                                                    (short)0, s, false, false);
      }
      #pragma unroll
      for (int r = 0; r < 8; ++r) {
        const float e = exp2_native(s[r]);   // Q pre-scaled by log2(e)/sqrt(D)
        sumexp += e;
        store_bf16_hi(&p_lds[w][(r + 8 * hi) * KC + f * 16 + ln], e);
      }
    }

    Frag16 pa;
    pa.h[0] = *(const bf16x8*)&p_lds[w][ln * KC + hi * 8];
    pa.h[1] = *(const bf16x8*)&p_lds[w][ln * KC + hi * 8 + 16];
    #pragma unroll
    for (int g = 0; g < 8; ++g) {
      Frag16 vb;
      const bf16x8* vp = (const bf16x8*)&vt_lds[(g * 16 + ln) * KC + hi * 8];
      vb.h[0] = vp[0]; vb.h[1] = vp[2];
      o[g] = __builtin_amdgcn_wmma_f32_16x16x32_bf16(false, pa.v, false, vb.v,
                                                     (short)0, o[g], false, false);
    }
  }

  {
    const int mrow = q0 + w * 16 + 8 * hi;
    #pragma unroll
    for (int g = 0; g < 8; ++g)
      #pragma unroll
      for (int r = 0; r < 8; ++r)
        out[bbase + (size_t)(mrow + r) * D_ + g * 16 + ln] = o[g][r];
  }

  #pragma unroll
  for (int off = 16; off > 0; off >>= 1)
    sumexp += __shfl_xor(sumexp, off);
  if (lane == 0) atomicAdd(total, sumexp);
}

// ---------------------------------------------------------------------------
// Pre-conversion kernels: K -> bf16 row-major; V -> bf16 transposed [B][D][S]
// ---------------------------------------------------------------------------
__global__ void cvt_k_bf16(const float* __restrict__ k, unsigned short* __restrict__ okb) {
  const size_t idx = (size_t)blockIdx.x * blockDim.x + threadIdx.x;
  const size_t n4 = (size_t)B_ * S_ * D_ / 4;
  if (idx < n4) {
    const float4 f = ((const float4*)k)[idx];
    union { __bf16 b[4]; uint2 u; } p;
    p.b[0] = f2bf(f.x); p.b[1] = f2bf(f.y); p.b[2] = f2bf(f.z); p.b[3] = f2bf(f.w);
    ((uint2*)okb)[idx] = p.u;
  }
}

__global__ void __launch_bounds__(256)
cvt_v_t(const float* __restrict__ v, unsigned short* __restrict__ ovt) {
  __shared__ __bf16 t[64 * 132];   // 64 keys x 128 d, padded stride
  const int tid = threadIdx.x;
  const int b   = blockIdx.y;
  const int k0  = blockIdx.x * 64;
  const size_t bbase = (size_t)b * S_ * D_;

  for (int e = tid; e < 64 * D_; e += 256) {
    const int key = e >> 7, d = e & (D_ - 1);
    t[key * 132 + d] = f2bf(v[bbase + (size_t)(k0 + key) * D_ + d]);
  }
  __syncthreads();
  for (int e = tid; e < D_ * 32; e += 256) {
    const int d = e >> 5, kp = e & 31;
    union { __bf16 b[2]; unsigned u; } p;
    p.b[0] = t[(2 * kp) * 132 + d];
    p.b[1] = t[(2 * kp + 1) * 132 + d];
    *(unsigned*)&ovt[(size_t)b * D_ * S_ + (size_t)d * S_ + k0 + 2 * kp] = p.u;
  }
}

// ---------------------------------------------------------------------------
// Main kernel: bf16 K / V^T from workspace, KC2=64 keys per iteration,
// double-buffered async global->LDS staging overlapped with WMMA compute.
// ---------------------------------------------------------------------------
__device__ __forceinline__ void stage_chunk(const __bf16* __restrict__ kb,
                                            const __bf16* __restrict__ vtb,
                                            size_t kelem, size_t vbase, int k0,
                                            __bf16* kdst, __bf16* vdst, int tid) {
  const __bf16* gk = kb + kelem + (size_t)k0 * D_;   // contiguous 16 KB
  for (int e = tid; e < (KC2 * D_) / 8; e += 256)
    cp16(gk + e * 8, kdst + e * 8);
  for (int e = tid; e < (D_ * KC2) / 8; e += 256) {
    const int d = e >> 3, j = e & 7;
    cp16(vtb + vbase + (size_t)d * S_ + k0 + j * 8, vdst + d * KC2 + j * 8);
  }
}

__global__ void __launch_bounds__(256)
attn_pass1_bf16(const float* __restrict__ q, const unsigned short* __restrict__ kb_,
                const unsigned short* __restrict__ vtb_, float* __restrict__ out,
                float* __restrict__ total) {
  __shared__ __align__(16) __bf16 q_lds[QT * D_];           // 32 KB
  __shared__ __align__(16) __bf16 k_lds[2][KC2 * D_];       // 2 x 16 KB [key][d]
  __shared__ __align__(16) __bf16 vt_lds[2][D_ * KC2];      // 2 x 16 KB [d][key]
  __shared__ __align__(16) __bf16 p_lds[NW][16 * KC2];      // 16 KB per-wave P

  const __bf16* kb  = (const __bf16*)kb_;
  const __bf16* vtb = (const __bf16*)vtb_;

  const int tid  = threadIdx.x;
  const int w    = tid >> 5;
  const int lane = tid & 31;
  const int ln   = lane & 15;
  const int hi   = lane >> 4;

  const int b  = blockIdx.y;
  const int q0 = blockIdx.x * QT;
  const size_t bbase = (size_t)b * S_ * D_;   // element base for q/out/kb
  const size_t vbase = (size_t)b * D_ * S_;   // element base for vtb

  // Stage first chunk while we convert Q
  stage_chunk(kb, vtb, bbase, vbase, 0, k_lds[0], vt_lds[0], tid);

  for (int e = tid; e < QT * D_; e += 256) {
    const int row = e >> 7, col = e & (D_ - 1);
    q_lds[e] = f2bf(q[bbase + (size_t)(q0 + row) * D_ + col] * QSCALE);
  }
  __syncthreads();

  Frag16 qa[4];
  {
    const int m = w * 16 + ln;
    #pragma unroll
    for (int c = 0; c < 4; ++c) {
      const int kbo = c * 32 + hi * 8;
      qa[c].h[0] = *(const bf16x8*)&q_lds[m * D_ + kbo];
      qa[c].h[1] = *(const bf16x8*)&q_lds[m * D_ + kbo + 16];
    }
  }

  v8f o[8];
  #pragma unroll
  for (int g = 0; g < 8; ++g) o[g] = (v8f)(0.0f);
  float sumexp = 0.0f;

  int buf = 0;
  for (int k0 = 0; k0 < S_; k0 += KC2) {
    wait_async0();        // this wave's copies into k_lds[buf]/vt_lds[buf] done
    __syncthreads();      // everyone's copies done; prev readers of buf^1 done

    if (k0 + KC2 < S_)    // overlap next chunk's DMA with this chunk's WMMAs
      stage_chunk(kb, vtb, bbase, vbase, k0 + KC2, k_lds[buf ^ 1], vt_lds[buf ^ 1], tid);

    const __bf16* kl = k_lds[buf];
    const __bf16* vl = vt_lds[buf];

    // S = Q*K^T : four 16x16 key fragments (4 independent WMMA chains)
    #pragma unroll
    for (int f = 0; f < 4; ++f) {
      v8f s = (v8f)(0.0f);
      const int krow = f * 16 + ln;
      #pragma unroll
      for (int c = 0; c < 4; ++c) {
        Frag16 kf;
        const bf16x8* kp = (const bf16x8*)&kl[krow * D_ + c * 32 + hi * 8];
        kf.h[0] = kp[0]; kf.h[1] = kp[2];
        s = __builtin_amdgcn_wmma_f32_16x16x32_bf16(false, qa[c].v, false, kf.v,
                                                    (short)0, s, false, false);
      }
      #pragma unroll
      for (int r = 0; r < 8; ++r) {
        const float e = exp2_native(s[r]);   // Q pre-scaled by log2(e)/sqrt(D)
        sumexp += e;
        store_bf16_hi(&p_lds[w][(r + 8 * hi) * KC2 + f * 16 + ln], e);
      }
    }

    // O += P * V : two K=32 key-halves x eight 16-wide d-groups
    #pragma unroll
    for (int pc = 0; pc < 2; ++pc) {
      Frag16 pa;
      const int pb = ln * KC2 + pc * 32 + hi * 8;
      pa.h[0] = *(const bf16x8*)&p_lds[w][pb];
      pa.h[1] = *(const bf16x8*)&p_lds[w][pb + 16];
      #pragma unroll
      for (int g = 0; g < 8; ++g) {
        Frag16 vb;
        const bf16x8* vp = (const bf16x8*)&vl[(g * 16 + ln) * KC2 + pc * 32 + hi * 8];
        vb.h[0] = vp[0]; vb.h[1] = vp[2];
        o[g] = __builtin_amdgcn_wmma_f32_16x16x32_bf16(false, pa.v, false, vb.v,
                                                       (short)0, o[g], false, false);
      }
    }

    __syncthreads();      // all readers done with buf before it is restaged
    buf ^= 1;
  }

  {
    const int mrow = q0 + w * 16 + 8 * hi;
    #pragma unroll
    for (int g = 0; g < 8; ++g)
      #pragma unroll
      for (int r = 0; r < 8; ++r)
        out[bbase + (size_t)(mrow + r) * D_ + g * 16 + ln] = o[g][r];
  }

  #pragma unroll
  for (int off = 16; off > 0; off >>= 1)
    sumexp += __shfl_xor(sumexp, off);
  if (lane == 0) atomicAdd(total, sumexp);
}

__global__ void zero_total(float* total) {
  if (threadIdx.x == 0 && blockIdx.x == 0) *total = 0.0f;
}

__global__ void scale_out(float* __restrict__ out, const float* __restrict__ total, int n) {
  const int i = blockIdx.x * blockDim.x + threadIdx.x;
  const float inv = 1.0f / *total;
  if (i < n) out[i] *= inv;
}

extern "C" void kernel_launch(void* const* d_in, const int* in_sizes, int n_in,
                              void* d_out, int out_size, void* d_ws, size_t ws_size,
                              hipStream_t stream) {
  const float* q = (const float*)d_in[0];
  const float* k = (const float*)d_in[1];
  const float* v = (const float*)d_in[2];
  float* out   = (float*)d_out;
  float* total = (float*)d_ws;

  zero_total<<<1, 64, 0, stream>>>(total);

  const size_t half = (size_t)B_ * S_ * D_ * 2;   // 8 MiB per bf16 array
  const size_t need = 256 + 2 * half;
  if (ws_size >= need) {
    unsigned short* kbws  = (unsigned short*)((char*)d_ws + 256);
    unsigned short* vtbws = (unsigned short*)((char*)d_ws + 256 + half);
    const int n4 = B_ * S_ * D_ / 4;
    cvt_k_bf16<<<(n4 + 255) / 256, 256, 0, stream>>>(k, kbws);
    cvt_v_t<<<dim3(S_ / 64, B_), 256, 0, stream>>>(v, vtbws);
    attn_pass1_bf16<<<dim3(S_ / QT, B_), 256, 0, stream>>>(q, kbws, vtbws, out, total);
  } else {
    attn_pass1<<<dim3(S_ / QT, B_), 256, 0, stream>>>(q, k, v, out, total);
  }

  const int n = B_ * S_ * D_;
  scale_out<<<(n + 255) / 256, 256, 0, stream>>>(out, total, n);
}
